// Qwen3NextSparseMoeBlock_618475290760
// MI455X (gfx1250) — matrix operand order repaired and checked by reading, not compiled, not run on hardware
//
#include <hip/hip_runtime.h>

// ---------------- problem constants ----------------
constexpr int kH  = 2048;         // hidden
constexpr int kT  = 2048;         // tokens (B*S)
constexpr int kE  = 32;           // routed experts
constexpr int kI  = 512;          // expert intermediate (== shared IS)
constexpr int kNE = 33;           // routed + shared expert
constexpr int kKT = kNE * kI;     // 16896 folded K dim for down-proj
constexpr int kKS_UP   = kH  / 32;   // 64  k-steps per expert (up/gate)
constexpr int kKS_DOWN = kKT / 32;   // 528 k-steps (down)

// ---------------- types ----------------
typedef __attribute__((ext_vector_type(16))) __bf16       v16bf;
typedef __attribute__((ext_vector_type(8)))  float        v8f;
typedef __attribute__((ext_vector_type(4)))  unsigned int v4u;

union FragAB {
  v16bf v;
  v4u   q[2];
  unsigned short s[16];
  unsigned int   u[8];
};

__device__ __forceinline__ unsigned short f2bfu(float f) {
  union { float f; unsigned u; } a; a.f = f;
  unsigned r = a.u + 0x7FFFu + ((a.u >> 16) & 1u);   // RNE
  return (unsigned short)(r >> 16);
}

// =====================================================================
// Kernel 1: router. 1 wave32 per token, lane == expert.
// =====================================================================
__global__ __launch_bounds__(256) void router_kernel(
    const float* __restrict__ x, const float* __restrict__ Wr,
    const float* __restrict__ Wsgate,
    float* __restrict__ Cext, float* __restrict__ rlogits) {
  const int w    = threadIdx.x >> 5;
  const int lane = threadIdx.x & 31;
  const int t    = blockIdx.x * 8 + w;
  if (t >= kT) return;

  const float* xr = x  + (size_t)t * kH;
  const float* wr = Wr + (size_t)lane * kH;

  float acc = 0.f;
  for (int h = 0; h < kH; h += 4) {
    float4 xv = *(const float4*)(xr + h);
    float4 wv = *(const float4*)(wr + h);
    acc += xv.x * wv.x + xv.y * wv.y + xv.z * wv.z + xv.w * wv.w;
  }
  float gacc = 0.f;
  for (int h = lane; h < kH; h += 32) gacc += xr[h] * Wsgate[h];

  rlogits[t * kE + lane] = acc;

  float m = acc;
  for (int off = 16; off; off >>= 1) m = fmaxf(m, __shfl_xor(m, off, 32));
  float p = __expf(acc - m);
  float s = p;
  for (int off = 16; off; off >>= 1) s += __shfl_xor(s, off, 32);
  const float rw = p / s;

  float v = rw, sum4 = 0.f;
  int selected = 0;
  for (int rep = 0; rep < 4; ++rep) {
    float bv = v; int bi = lane;
    for (int off = 16; off; off >>= 1) {
      float ov = __shfl_xor(bv, off, 32);
      int   oi = __shfl_xor(bi, off, 32);
      if (ov > bv || (ov == bv && oi < bi)) { bv = ov; bi = oi; }
    }
    sum4 += bv;
    if (lane == bi) { selected = 1; v = -1.f; }
  }
  Cext[t * kNE + lane] = selected ? (rw / sum4) : 0.f;

  for (int off = 16; off; off >>= 1) gacc += __shfl_xor(gacc, off, 32);
  if (lane == 0) Cext[t * kNE + 32] = 1.f / (1.f + __expf(-gacc));
}

// =====================================================================
// Kernel 2: fp32 -> bf16 of hidden states (natural [T][H] layout)
// =====================================================================
__global__ __launch_bounds__(256) void cvt_kernel(
    const float* __restrict__ x, unsigned short* __restrict__ xbf) {
  size_t i = ((size_t)blockIdx.x * 256 + threadIdx.x) * 4;
  if (i >= (size_t)kT * kH) return;
  float4 vv = *(const float4*)(x + i);
  ushort4 o;
  o.x = f2bfu(vv.x); o.y = f2bfu(vv.y); o.z = f2bfu(vv.z); o.w = f2bfu(vv.w);
  *(ushort4*)(xbf + i) = o;
}

// =====================================================================
// Kernel 3: pack gate/up weights into per-lane WMMA B-fragment layout.
//   dst[((e*32 + it)*64 + ks)*512 + lane*16 + j], lane=(hi<<4)|n:
//   column i = it*16 + n, elements k = ks*32 + hi*16 + (0..15).
// =====================================================================
__global__ __launch_bounds__(256) void pack_gu_kernel(
    const float* __restrict__ W, const float* __restrict__ Wsh,
    unsigned short* __restrict__ dst) {
  const size_t slot = (size_t)blockIdx.x * 256 + threadIdx.x;  // < 33*32*64*32
  const int lane = slot & 31;
  const size_t blk = slot >> 5;        // (e*32 + it)*64 + ks
  const int ks = blk % kKS_UP;
  const size_t eit = blk / kKS_UP;
  const int it = eit % 32;
  const int e  = eit / 32;
  const int i  = it * 16 + (lane & 15);
  const int k0 = ks * 32 + (lane >> 4) * 16;

  const float* src = ((e < kE) ? (W + (size_t)e * kI * kH) : Wsh)
                     + (size_t)i * kH + k0;
  unsigned short* d = dst + slot * 16;
#pragma unroll
  for (int q = 0; q < 4; ++q) {
    float4 vv = *(const float4*)(src + q * 4);
    d[q * 4 + 0] = f2bfu(vv.x); d[q * 4 + 1] = f2bfu(vv.y);
    d[q * 4 + 2] = f2bfu(vv.z); d[q * 4 + 3] = f2bfu(vv.w);
  }
}

// =====================================================================
// Kernel 4: pack down-proj weights (folded 33-expert K) the same way.
// =====================================================================
__global__ __launch_bounds__(256) void pack_down_kernel(
    const float* __restrict__ Wd, const float* __restrict__ Wsd,
    unsigned short* __restrict__ dst) {
  const size_t slot = (size_t)blockIdx.x * 256 + threadIdx.x;  // < 128*528*32
  const int lane = slot & 31;
  const size_t blk = slot >> 5;        // ht*528 + ks
  const int ks = blk % kKS_DOWN;
  const int ht = blk / kKS_DOWN;
  const int h  = ht * 16 + (lane & 15);
  const int kf = ks * 32 + (lane >> 4) * 16;
  const int e  = kf >> 9;
  const int i  = kf & 511;

  const float* src = ((e < kE) ? (Wd + (size_t)e * kH * kI) : Wsd)
                     + (size_t)h * kI + i;
  unsigned short* d = dst + slot * 16;
#pragma unroll
  for (int q = 0; q < 4; ++q) {
    float4 vv = *(const float4*)(src + q * 4);
    d[q * 4 + 0] = f2bfu(vv.x); d[q * 4 + 1] = f2bfu(vv.y);
    d[q * 4 + 2] = f2bfu(vv.z); d[q * 4 + 3] = f2bfu(vv.w);
  }
}

// ---------------------------------------------------------------------
// Async double-buffered A-panel staging (CDNA5 GLOBAL_LOAD_ASYNC_TO_LDS).
// Panel layout (per buffer): apanel[buf][mtile(4)][kstep(2)][lane(32)][8u].
// ---------------------------------------------------------------------
#define PANEL_UINTS (4 * 2 * 32 * 8)         // 2048 uints = 8 KB per buffer
#define PANEL_BYTES (PANEL_UINTS * 4)

__device__ __forceinline__ void async_issue_panel(
    unsigned ldsOff0, int buf,
    const unsigned short* __restrict__ g0, size_t rowStride /*shorts*/) {
  const unsigned l0 = ldsOff0 + buf * PANEL_BYTES;        // mtile stride 2KB
  const unsigned short* g1 = g0 + 16 * rowStride;
  const unsigned short* g2 = g1 + 16 * rowStride;
  const unsigned short* g3 = g2 + 16 * rowStride;
  asm volatile(
      "global_load_async_to_lds_b64 %0, %4, off\n\t"
      "global_load_async_to_lds_b64 %1, %5, off\n\t"
      "global_load_async_to_lds_b64 %2, %6, off\n\t"
      "global_load_async_to_lds_b64 %3, %7, off"
      :: "v"(l0), "v"(l0 + 2048u), "v"(l0 + 4096u), "v"(l0 + 6144u),
         "v"(g0), "v"(g1), "v"(g2), "v"(g3)
      : "memory");
}

// =====================================================================
// Kernel 5: up/gate WMMA GEMM for 33 "experts" (32 routed + shared).
// grid = (kI/128, kT/64, 33); 8 waves; wave w owns n-tile it=bx*8+w and
// ALL 4 m-tiles (register B reuse): per k-step 1x(Bg,Bu) fetch -> 8 WMMA.
// =====================================================================
__global__ __launch_bounds__(256) void moe_up_kernel(
    const unsigned short* __restrict__ xbf,
    const unsigned short* __restrict__ Wgp,
    const unsigned short* __restrict__ Wup,
    const float* __restrict__ Cext,
    unsigned short* __restrict__ Abf) {
  const int e    = blockIdx.z;
  const int t0   = blockIdx.y * 64;
  const int tid  = threadIdx.x;
  const int w    = tid >> 5, lane = tid & 31;
  const int mrow = lane & 15, hi = lane >> 4;
  const int it   = blockIdx.x * 8 + w;          // 16-wide i tile per wave

  __shared__ unsigned int apanel[2][4][2][32][8];   // 16 KB

  // per-thread async staging map (affine, computed once)
  const int trow0 = tid >> 4;                 // 0..15
  const int kq    = (tid & 15) << 2;          // short offset in 64-panel
  const int skst  = kq >> 5;
  const int kl    = kq & 31;
  const int sg    = kl >> 3;
  const int slane = (kl & 16) | (trow0 & 15);
  const int sui   = (kl & 15) >> 1;
  const unsigned ldsBase = (unsigned)(size_t)(const void*)&apanel[0][0][0][0][0];
  const unsigned ldsOff0 = ldsBase + (unsigned)(((skst * 32 + slane) * 8 + sui) * 4);
  const unsigned short* gbase = xbf + (size_t)(t0 + trow0) * kH + kq;

  v8f accg[4] = {}, accu[4] = {};

  const size_t bstep = 512;                   // shorts per k-step
  const unsigned short* bg = Wgp + ((size_t)(e * 32 + it) * kKS_UP) * bstep + lane * 16;
  const unsigned short* bu = Wup + ((size_t)(e * 32 + it) * kKS_UP) * bstep + lane * 16;

  constexpr int NP = kH / 64;                 // 32 panels
  async_issue_panel(ldsOff0, 0, gbase, kH);

  for (int p = 0; p < NP; ++p) {
    const int b = p & 1;
    if (p + 1 < NP) {
      async_issue_panel(ldsOff0, 1 - b, gbase + (size_t)(p + 1) * 64, kH);
      asm volatile("s_wait_asynccnt 0x4" ::: "memory");
    } else {
      asm volatile("s_wait_asynccnt 0x0" ::: "memory");
    }
    __syncthreads();

#pragma unroll
    for (int kstep = 0; kstep < 2; ++kstep) {
      FragAB fg, fu;
      const v4u* gp = (const v4u*)(bg + kstep * bstep);
      const v4u* up = (const v4u*)(bu + kstep * bstep);
      fg.q[0] = gp[0]; fg.q[1] = gp[1];
      fu.q[0] = up[0]; fu.q[1] = up[1];

#pragma unroll
      for (int m = 0; m < 4; ++m) {
        FragAB fa;
        const v4u* ap = (const v4u*)&apanel[b][m][kstep][lane][0];
        fa.q[0] = ap[0]; fa.q[1] = ap[1];
        accg[m] = __builtin_amdgcn_wmma_f32_16x16x32_bf16(
            false, fa.v, false, fg.v, (short)0, accg[m], false, false);
        accu[m] = __builtin_amdgcn_wmma_f32_16x16x32_bf16(
            false, fa.v, false, fu.v, (short)0, accu[m], false, false);
      }
    }
    bg += 2 * bstep; bu += 2 * bstep;
    __syncthreads();
  }

  const int ncol = it * 16 + mrow;
#pragma unroll
  for (int m = 0; m < 4; ++m) {
#pragma unroll
    for (int r = 0; r < 8; ++r) {
      const int t = t0 + m * 16 + r + hi * 8;
      const float c = Cext[t * kNE + e];
      const float g = accg[m][r], u = accu[m][r];
      const float a = (g / (1.f + __expf(-g))) * u * c;
      Abf[(size_t)t * kKT + e * kI + ncol] = f2bfu(a);
    }
  }
}

// =====================================================================
// Kernel 6: folded down-projection out[T,H] = Abf[T,16896] @ Bdp.
// grid = (kH/128, kT/64); wave w owns h-tile ht=bx*8+w and all 4 m-tiles.
// =====================================================================
__global__ __launch_bounds__(256) void moe_down_kernel(
    const unsigned short* __restrict__ Abf,
    const unsigned short* __restrict__ Bdp,
    float* __restrict__ out) {
  const int t0   = blockIdx.y * 64;
  const int tid  = threadIdx.x;
  const int w    = tid >> 5, lane = tid & 31;
  const int mrow = lane & 15, hi = lane >> 4;
  const int ht   = blockIdx.x * 8 + w;          // 16-wide h tile per wave

  __shared__ unsigned int apanel[2][4][2][32][8];   // 16 KB

  const int trow0 = tid >> 4;
  const int kq    = (tid & 15) << 2;
  const int skst  = kq >> 5;
  const int kl    = kq & 31;
  const int sg    = kl >> 3;
  const int slane = (kl & 16) | (trow0 & 15);
  const int sui   = (kl & 15) >> 1;
  const unsigned ldsBase = (unsigned)(size_t)(const void*)&apanel[0][0][0][0][0];
  const unsigned ldsOff0 = ldsBase + (unsigned)(((skst * 32 + slane) * 8 + sui) * 4);
  const unsigned short* gbase = Abf + (size_t)(t0 + trow0) * kKT + kq;

  v8f acc[4] = {};

  const size_t bstep = 512;
  const unsigned short* bd = Bdp + ((size_t)ht * kKS_DOWN) * bstep + lane * 16;

  constexpr int NP = kKT / 64;                // 264 panels
  async_issue_panel(ldsOff0, 0, gbase, kKT);

  for (int p = 0; p < NP; ++p) {
    const int b = p & 1;
    if (p + 1 < NP) {
      async_issue_panel(ldsOff0, 1 - b, gbase + (size_t)(p + 1) * 64, kKT);
      asm volatile("s_wait_asynccnt 0x4" ::: "memory");
    } else {
      asm volatile("s_wait_asynccnt 0x0" ::: "memory");
    }
    __syncthreads();

#pragma unroll
    for (int kstep = 0; kstep < 2; ++kstep) {
      FragAB fb;
      const v4u* bp = (const v4u*)(bd + kstep * bstep);
      fb.q[0] = bp[0]; fb.q[1] = bp[1];

#pragma unroll
      for (int m = 0; m < 4; ++m) {
        FragAB fa;
        const v4u* ap = (const v4u*)&apanel[b][m][kstep][lane][0];
        fa.q[0] = ap[0]; fa.q[1] = ap[1];
        acc[m] = __builtin_amdgcn_wmma_f32_16x16x32_bf16(
            false, fa.v, false, fb.v, (short)0, acc[m], false, false);
      }
    }
    bd += 2 * bstep;
    __syncthreads();
  }

  const int hcol = ht * 16 + mrow;
#pragma unroll
  for (int m = 0; m < 4; ++m) {
#pragma unroll
    for (int r = 0; r < 8; ++r) {
      const int t = t0 + m * 16 + r + hi * 8;
      out[(size_t)t * kH + hcol] = acc[m][r];
    }
  }
}

// =====================================================================
// host launch
// =====================================================================
extern "C" void kernel_launch(void* const* d_in, const int* in_sizes, int n_in,
                              void* d_out, int out_size, void* d_ws, size_t ws_size,
                              hipStream_t stream) {
  (void)in_sizes; (void)n_in; (void)out_size; (void)ws_size;
  const float* x      = (const float*)d_in[0];
  const float* Wr     = (const float*)d_in[1];
  const float* Wg     = (const float*)d_in[2];
  const float* Wu     = (const float*)d_in[3];
  const float* Wd     = (const float*)d_in[4];
  const float* Wsg    = (const float*)d_in[5];
  const float* Wsu    = (const float*)d_in[6];
  const float* Wsd    = (const float*)d_in[7];
  const float* Wsgate = (const float*)d_in[8];

  float* out = (float*)d_out;
  float* rl  = out + (size_t)kT * kH;           // router_logits tail

  char* ws = (char*)d_ws;
  size_t off = 0;
  auto take = [&](size_t bytes) {
    char* p = ws + off;
    off += (bytes + 255) & ~(size_t)255;
    return p;
  };
  const size_t wpackElems = (size_t)kNE * kI * kH;    // 33*512*2048
  float*          Cext = (float*)         take((size_t)kT * kNE * 4);
  unsigned short* xbf  = (unsigned short*)take((size_t)kT * kH * 2);
  unsigned short* Abf  = (unsigned short*)take((size_t)kT * kKT * 2);
  unsigned short* Wgp  = (unsigned short*)take(wpackElems * 2);
  unsigned short* Wup  = (unsigned short*)take(wpackElems * 2);
  unsigned short* Bdp  = (unsigned short*)take(wpackElems * 2);

  router_kernel<<<kT / 8, 256, 0, stream>>>(x, Wr, Wsgate, Cext, rl);
  cvt_kernel<<<((size_t)kT * kH / 4 + 255) / 256, 256, 0, stream>>>(x, xbf);

  const int packBlocks = (int)(wpackElems / 16 / 256);   // 8448
  pack_gu_kernel<<<packBlocks, 256, 0, stream>>>(Wg, Wsg, Wgp);
  pack_gu_kernel<<<packBlocks, 256, 0, stream>>>(Wu, Wsu, Wup);
  pack_down_kernel<<<packBlocks, 256, 0, stream>>>(Wd, Wsd, Bdp);

  dim3 g3(kI / 128, kT / 64, kNE);
  moe_up_kernel<<<g3, 256, 0, stream>>>(xbf, Wgp, Wup, Cext, Abf);

  dim3 g4(kH / 128, kT / 64);
  moe_down_kernel<<<g4, 256, 0, stream>>>(Abf, Bdp, out);
}
